// Chebyshev_64716567216739
// MI455X (gfx1250) — matrix operand order repaired and checked by reading, not compiled, not run on hardware
//
#include <hip/hip_runtime.h>
#include <stdint.h>

#define CB    4
#define CM    50000
#define CFIN  128
#define CFOUT 128
#define CK    4
#define CE    800000
#define CW    512              // FIN*B == FIN*K == 512
#define NROWS (CB * CM)        // 200000
#define NTILES (NROWS / 16)    // 12500
#define NJOBS  (NTILES / 2)    // 6250 (two 16-row tiles per wave)

typedef __attribute__((ext_vector_type(16))) __bf16 v16bf;
typedef __attribute__((ext_vector_type(8)))  float  v8f;
typedef int v4i_t __attribute__((vector_size(16)));

union ABFrag {
  v16bf v;
  float4 f4[2];
};

__device__ __forceinline__ unsigned short f2bf(float f) {
  unsigned int u = __float_as_uint(f);
  u += 0x7FFFu + ((u >> 16) & 1u);   // round-to-nearest-even
  return (unsigned short)(u >> 16);
}

// ---------------------------------------------------------------- zero init
__global__ void zero_kernel(int* __restrict__ cnt, float* __restrict__ ssum) {
  int i = blockIdx.x * 256 + threadIdx.x;
  if (i < CM) cnt[i] = 0;
  if (i < CB * CFOUT) ssum[i] = 0.0f;
}

// ---------------------------------------------------------------- CSR build
__global__ void count_kernel(const int* __restrict__ rows, int* __restrict__ cnt) {
  int e = blockIdx.x * 256 + threadIdx.x;
  if (e < CE) atomicAdd(&cnt[rows[e]], 1);
}

__global__ void scan_kernel(int* cnt_cursor, int* __restrict__ rowptr) {
  __shared__ int sh[256];
  __shared__ int run;
  const int tid = threadIdx.x;
  if (tid == 0) run = 0;
  __syncthreads();
  for (int base = 0; base < CM; base += 256) {
    const int i = base + tid;
    int v = (i < CM) ? cnt_cursor[i] : 0;
    sh[tid] = v;
    __syncthreads();
    for (int off = 1; off < 256; off <<= 1) {
      int add = (tid >= off) ? sh[tid - off] : 0;
      __syncthreads();
      sh[tid] += add;
      __syncthreads();
    }
    const int excl = sh[tid] - v;
    const int rp = run + excl;
    if (i < CM) { rowptr[i] = rp; cnt_cursor[i] = rp; }  // cursor reuses cnt buffer
    __syncthreads();
    if (tid == 0) run += sh[255];
    __syncthreads();
  }
  if (tid == 0) rowptr[CM] = run;
}

__global__ void scatter_kernel(const int* __restrict__ rows,
                               const int* __restrict__ cols,
                               const float* __restrict__ vals,
                               int* __restrict__ cursor,
                               float* __restrict__ evals,
                               int* __restrict__ ecols) {
  int e = blockIdx.x * 256 + threadIdx.x;
  if (e >= CE) return;
  int r = rows[e];
  int p = atomicAdd(&cursor[r], 1);
  evals[p] = vals[e];
  ecols[p] = cols[e];
}

// ------------------------------------------------- x -> x0 [M, fin*4+b] + XS(k=0)
__global__ void init_x0_kernel(const float* __restrict__ x,
                               float* __restrict__ xt,
                               unsigned short* __restrict__ XS) {
  int id = blockIdx.x * 256 + threadIdx.x;     // CM*CFIN threads
  if (id >= CM * CFIN) return;
  int m = id >> 7;
  int fin = id & 127;
#pragma unroll
  for (int b = 0; b < CB; ++b) {
    float v = x[((size_t)b * CM + m) * CFIN + fin];
    xt[(size_t)m * CW + fin * 4 + b] = v;
    XS[(((size_t)b * CM + m) << 9) + (fin << 2) + 0] = f2bf(v);
  }
}

// ------------------------------------------------- kernel -> swizzled B-fragments
// Block (kb,n): 1 KB; lane's V0..V3 halves in part0 (lane*16B), V4..V7 in part1.
// 16-bit B 32x16 layout: lanes 0-15 hold K=0..7,16..23 (col=lane), lanes 16-31 K=8..15,24..31.
__global__ void kswz_kernel(const float* __restrict__ kmat,
                            unsigned short* __restrict__ kswz) {
  int id = blockIdx.x * 256 + threadIdx.x;     // 65536 threads
  if (id >= 16 * 8 * 32 * 16) return;
  int e    = id & 15;
  int lane = (id >> 4) & 31;
  int n    = (id >> 9) & 7;
  int kb   = id >> 12;
  int kr = kb * 32 + ((e >> 3) << 4) + ((lane >> 4) << 3) + (e & 7);
  float v = kmat[kr * CFOUT + n * 16 + (lane & 15)];
  kswz[(kb * 8 + n) * 512 + ((e >> 3) << 8) + lane * 8 + (e & 7)] = f2bf(v);
}

// ------------------------------------------------- SpMM: one wave32 per row
// Xout[m,:] = alpha * (L @ Xin)[m,:] + beta * Xprev[m,:]; also writes bf16 XS slice k.
__global__ void spmm_kernel(const int* __restrict__ rowptr,
                            const float* __restrict__ evals,
                            const int* __restrict__ ecols,
                            const float* __restrict__ Xin,
                            const float* __restrict__ Xprev,
                            float* __restrict__ Xout,
                            unsigned short* __restrict__ XS,
                            int kidx, float alpha, float beta) {
  const int wave = threadIdx.x >> 5;
  const int lane = threadIdx.x & 31;
  const int m = blockIdx.x * 8 + wave;
  if (m >= CM) return;

  float4 acc[4] = {};
  const int start = rowptr[m];
  const int end   = rowptr[m + 1];
  for (int base = start; base < end; base += 32) {
    const int rem = end - base;
    float v = 0.0f; int c = 0;
    if (lane < rem) { v = evals[base + lane]; c = ecols[base + lane]; }
    const int cnt = rem < 32 ? rem : 32;
    for (int j = 0; j < cnt; ++j) {
      const float vj = __shfl(v, j);
      const int   cj = __shfl(c, j);
      const float4* src = (const float4*)(Xin + (size_t)cj * CW);
#pragma unroll
      for (int i = 0; i < 4; ++i) {
        float4 s = src[i * 32 + lane];
        acc[i].x = fmaf(vj, s.x, acc[i].x);
        acc[i].y = fmaf(vj, s.y, acc[i].y);
        acc[i].z = fmaf(vj, s.z, acc[i].z);
        acc[i].w = fmaf(vj, s.w, acc[i].w);
      }
    }
  }

  const float4* prevp = (const float4*)(Xprev + (size_t)m * CW);
  float4* outp = (float4*)(Xout + (size_t)m * CW);
#pragma unroll
  for (int i = 0; i < 4; ++i) {
    float4 o = acc[i];
    o.x *= alpha; o.y *= alpha; o.z *= alpha; o.w *= alpha;
    if (beta != 0.0f) {
      float4 p = prevp[i * 32 + lane];
      o.x = fmaf(beta, p.x, o.x);
      o.y = fmaf(beta, p.y, o.y);
      o.z = fmaf(beta, p.z, o.z);
      o.w = fmaf(beta, p.w, o.w);
    }
    outp[i * 32 + lane] = o;
    const int fin = i * 32 + lane;       // feature f = i*128+lane*4+q -> fin, b=q
    XS[(((size_t)0 * CM + m) << 9) + (fin << 2) + kidx] = f2bf(o.x);
    XS[(((size_t)1 * CM + m) << 9) + (fin << 2) + kidx] = f2bf(o.y);
    XS[(((size_t)2 * CM + m) << 9) + (fin << 2) + kidx] = f2bf(o.z);
    XS[(((size_t)3 * CM + m) << 9) + (fin << 2) + kidx] = f2bf(o.w);
  }
}

// ------------------------------------------------- GEMM (bf16 WMMA) + ReLU
// Each wave computes two 16-row tiles (32 rows x 128 cols), sharing B fragments.
// 256 threads, waves_per_eu(4): 256-VGPR cap (no spills, ~190 live), occupancy
// matches the 128KB-LDS limit (2 workgroups / WGP).
__global__ void
__launch_bounds__(256)
__attribute__((amdgpu_waves_per_eu(4)))
gemm_relu_kernel(const unsigned short* __restrict__ XS,
                 const unsigned short* __restrict__ kswz,
                 float* __restrict__ y) {
  extern __shared__ unsigned short ldsB[];      // 65536 halfwords = 128 KB
  const int tid = threadIdx.x;

  // Stage swizzled kernel matrix into LDS.
#if __has_builtin(__builtin_amdgcn_global_load_async_to_lds_b128)
  {
    v4i_t* g = (v4i_t*)kswz;
    v4i_t* l = (v4i_t*)ldsB;
#pragma unroll 1
    for (int i = 0; i < 32; ++i) {
      const int idx = i * 256 + tid;           // 16-byte units
      __builtin_amdgcn_global_load_async_to_lds_b128(g + idx, l + idx, 0, 0);
    }
  }
#if __has_builtin(__builtin_amdgcn_s_wait_asynccnt)
  __builtin_amdgcn_s_wait_asynccnt(0);
#else
  asm volatile("s_wait_asynccnt 0x0" ::: "memory");
#endif
#else
  {
    const float4* src = (const float4*)kswz;
    float4* dst = (float4*)ldsB;
    for (int i = tid; i < 65536 / 8; i += 256) dst[i] = src[i];
  }
#endif
  __syncthreads();

  const int wave = tid >> 5;
  const int lane = tid & 31;
  const int job = blockIdx.x * 8 + wave;
  if (job >= NJOBS) return;

  // A fragment bases: 16-bit A 16x32 layout; lane row = tile*16 + (lane&15),
  // lanes>=16 offset by 8 halfwords in K, V4..V7 at +16 halfwords.
  const unsigned short* arow0 = XS + (size_t)(job * 2) * (16 * CW)
                                   + (size_t)(lane & 15) * CW + ((lane >> 4) << 3);
  const unsigned short* arow1 = arow0 + 16 * CW;

  v8f acc0[8] = {};
  v8f acc1[8] = {};
  for (int kb = 0; kb < 16; ++kb) {
    ABFrag a0, a1;
    a0.f4[0] = *(const float4*)(arow0 + kb * 32);
    a0.f4[1] = *(const float4*)(arow0 + kb * 32 + 16);
    a1.f4[0] = *(const float4*)(arow1 + kb * 32);
    a1.f4[1] = *(const float4*)(arow1 + kb * 32 + 16);
    const unsigned short* lb = ldsB + kb * 4096 + lane * 8;
#pragma unroll
    for (int n = 0; n < 8; ++n) {
      ABFrag b;
      b.f4[0] = *(const float4*)(lb + n * 512);
      b.f4[1] = *(const float4*)(lb + n * 512 + 256);
      acc0[n] = __builtin_amdgcn_wmma_f32_16x16x32_bf16(
          false, a0.v, false, b.v, (short)0, acc0[n], false, false);
      acc1[n] = __builtin_amdgcn_wmma_f32_16x16x32_bf16(
          false, a1.v, false, b.v, (short)0, acc1[n], false, false);
    }
  }

  // C/D layout: VGPR r -> (M = r + (lane<16?0:8), N = lane&15)
  const int col = lane & 15;
  const int rbase = (lane >> 4) << 3;
  float* yt0 = y + (size_t)(job * 2) * (16 * CFOUT);
  float* yt1 = yt0 + 16 * CFOUT;
#pragma unroll
  for (int n = 0; n < 8; ++n)
#pragma unroll
    for (int r = 0; r < 8; ++r) {
      yt0[(rbase + r) * CFOUT + n * 16 + col] = fmaxf(acc0[n][r], 0.0f);
      yt1[(rbase + r) * CFOUT + n * 16 + col] = fmaxf(acc1[n][r], 0.0f);
    }
}

// ------------------------------------------------- SE block
__global__ void se_reduce_kernel(const float* __restrict__ y, float* __restrict__ ssum) {
  __shared__ float red[256];
  const int b = blockIdx.x >> 6;
  const int s = blockIdx.x & 63;
  const int tid = threadIdx.x;
  const int fo = tid & 127;
  const int half = tid >> 7;
  float sum = 0.0f;
  for (int m = s * 2 + half; m < CM; m += 128)
    sum += y[((size_t)b * CM + m) * CFOUT + fo];
  red[tid] = sum;
  __syncthreads();
  if (tid < 128) atomicAdd(&ssum[b * CFOUT + tid], red[tid] + red[tid + 128]);
}

__global__ void se_mlp_kernel(const float* __restrict__ ssum,
                              const float* __restrict__ Wd, const float* __restrict__ bd,
                              const float* __restrict__ Wu, const float* __restrict__ bu,
                              float* __restrict__ U) {
  __shared__ float ls[CFOUT];
  __shared__ float ld[16];
  const int b = blockIdx.x;
  const int fo = threadIdx.x;
  ls[fo] = ssum[b * CFOUT + fo] * (1.0f / (float)CM);
  __syncthreads();
  if (fo < 16) {
    float d = bd[fo];
    for (int j = 0; j < CFOUT; ++j) d = fmaf(ls[j], Wd[j * 16 + fo], d);
    ld[fo] = d / (1.0f + __expf(-d));           // swish
  }
  __syncthreads();
  float u = bu[fo];
  for (int h = 0; h < 16; ++h) u = fmaf(ld[h], Wu[h * CFOUT + fo], u);
  U[b * CFOUT + fo] = 1.0f / (1.0f + __expf(-u));   // sigmoid
}

__global__ void scale_kernel(float* __restrict__ y, const float* __restrict__ U) {
  const int id = blockIdx.x * 256 + threadIdx.x;     // NROWS*CFOUT/4 threads
  if (id >= NROWS * CFOUT / 4) return;
  const size_t flat = (size_t)id * 4;
  const int b = (int)(flat / ((size_t)CM * CFOUT));
  const int fo = (int)(flat & (CFOUT - 1));
  float4 v = ((float4*)y)[id];
  const float* u = U + b * CFOUT + fo;
  v.x *= u[0]; v.y *= u[1]; v.z *= u[2]; v.w *= u[3];
  ((float4*)y)[id] = v;
}

// ---------------------------------------------------------------- launcher
static inline char* align256(char* p) {
  return (char*)(((uintptr_t)p + 255) & ~(uintptr_t)255);
}

extern "C" void kernel_launch(void* const* d_in, const int* in_sizes, int n_in,
                              void* d_out, int out_size, void* d_ws, size_t ws_size,
                              hipStream_t stream) {
  const float* x      = (const float*)d_in[0];
  const float* Lvals  = (const float*)d_in[1];
  const int*   Lrows  = (const int*)  d_in[2];
  const int*   Lcols  = (const int*)  d_in[3];
  const float* kmat   = (const float*)d_in[4];
  const float* Wd     = (const float*)d_in[5];
  const float* bd     = (const float*)d_in[6];
  const float* Wu     = (const float*)d_in[7];
  const float* bu     = (const float*)d_in[8];
  float* y = (float*)d_out;

  char* p = (char*)d_ws;
  float* xtA = (float*)p;            p = align256(p + (size_t)CM * CW * 4);
  float* xtB = (float*)p;            p = align256(p + (size_t)CM * CW * 4);
  float* xtC = (float*)p;            p = align256(p + (size_t)CM * CW * 4);
  unsigned short* XS = (unsigned short*)p;   p = align256(p + (size_t)NROWS * CW * 2);
  unsigned short* kswz = (unsigned short*)p; p = align256(p + (size_t)65536 * 2);
  int* rowptr = (int*)p;             p = align256(p + (size_t)(CM + 1) * 4);
  int* cnt_cur = (int*)p;            p = align256(p + (size_t)CM * 4);
  float* evals = (float*)p;          p = align256(p + (size_t)CE * 4);
  int* ecols = (int*)p;              p = align256(p + (size_t)CE * 4);
  float* ssum = (float*)p;           p = align256(p + (size_t)CB * CFOUT * 4);
  float* U = (float*)p;              p = align256(p + (size_t)CB * CFOUT * 4);

  // CSR build (deterministic work; edge order within a row via atomics)
  zero_kernel<<<(CM + 255) / 256, 256, 0, stream>>>(cnt_cur, ssum);
  count_kernel<<<(CE + 255) / 256, 256, 0, stream>>>(Lrows, cnt_cur);
  scan_kernel<<<1, 256, 0, stream>>>(cnt_cur, rowptr);
  scatter_kernel<<<(CE + 255) / 256, 256, 0, stream>>>(Lrows, Lcols, Lvals,
                                                       cnt_cur, evals, ecols);

  // x -> x0 layout + bf16 XS slice k=0; kernel -> swizzled bf16 B-fragments
  init_x0_kernel<<<(CM * CFIN + 255) / 256, 256, 0, stream>>>(x, xtA, XS);
  kswz_kernel<<<(16 * 8 * 32 * 16 + 255) / 256, 256, 0, stream>>>(kmat, kswz);

  // Chebyshev recurrence: x1 = L@x0 ; x2 = 2 L@x1 - x0 ; x3 = 2 L@x2 - x1
  spmm_kernel<<<(CM + 7) / 8, 256, 0, stream>>>(rowptr, evals, ecols,
      xtA, xtA, xtB, XS, 1, 1.0f, 0.0f);
  spmm_kernel<<<(CM + 7) / 8, 256, 0, stream>>>(rowptr, evals, ecols,
      xtB, xtA, xtC, XS, 2, 2.0f, -1.0f);
  spmm_kernel<<<(CM + 7) / 8, 256, 0, stream>>>(rowptr, evals, ecols,
      xtC, xtB, xtA, XS, 3, 2.0f, -1.0f);

  // y = relu(xs @ kernel) via bf16 WMMA, 128 KB LDS-staged B-fragments
  gemm_relu_kernel<<<(NJOBS + 7) / 8, 256, 131072, stream>>>(XS, kswz, y);

  // SE block + in-place scale of d_out
  se_reduce_kernel<<<CB * 64, 256, 0, stream>>>(y, ssum);
  se_mlp_kernel<<<CB, 128, 0, stream>>>(ssum, Wd, bd, Wu, bu, U);
  scale_kernel<<<(NROWS * CFOUT / 4 + 255) / 256, 256, 0, stream>>>(y, U);

  (void)in_sizes; (void)n_in; (void)out_size; (void)ws_size;
}
